// QuantLinearLoRA_81389630259464
// MI455X (gfx1250) — matrix operand order repaired and checked by reading, not compiled, not run on hardware
//
#include <hip/hip_runtime.h>
#include <hip/hip_bf16.h>
#include <stdint.h>

typedef __attribute__((ext_vector_type(16))) __bf16 v16bf;
typedef __attribute__((ext_vector_type(8)))  __bf16 v8bf;
typedef __attribute__((ext_vector_type(8)))  float  v8f;

typedef unsigned int u32x4 __attribute__((ext_vector_type(4)));
typedef int          i32x4 __attribute__((ext_vector_type(4)));
typedef int          i32x8 __attribute__((ext_vector_type(8)));

#if __has_builtin(__builtin_amdgcn_tensor_load_to_lds)
#define ATHENA_HAVE_TDM 1
#else
#define ATHENA_HAVE_TDM 0
#endif

namespace {
constexpr int   kM     = 16384;   // 4 * 4096 rows
constexpr int   kK     = 4096;
constexpr int   kN     = 4096;
constexpr int   kR     = 16;      // LoRA rank
constexpr float kAlpha = 2.0f;
}

// ---------------------------------------------------------------------------
// Kernel 0 (TDM path only): one-time int8 -> bf16 conversion of the weights.
// ---------------------------------------------------------------------------
__launch_bounds__(256)
__global__ void convert_q_kernel(const int8_t* __restrict__ qw,
                                 __bf16* __restrict__ qb) {
  const size_t i = ((size_t)blockIdx.x * 256 + threadIdx.x) * 16;
  union { int4 v; int8_t b[16]; } u;
  u.v = *(const int4*)(qw + i);
  v8bf h0, h1;
#pragma unroll
  for (int j = 0; j < 8; ++j) {
    h0[j] = (__bf16)(float)u.b[j];        // int8 exact in bf16
    h1[j] = (__bf16)(float)u.b[8 + j];
  }
  *(v8bf*)(qb + i)     = h0;
  *(v8bf*)(qb + i + 8) = h1;
}

// ---------------------------------------------------------------------------
// Kernel 1: xa[m,r] = x[m,:] @ lora_A[:,r]  (fp32),
//           rs[m]   = sum_i bf16(x[m,i]),
//           optionally xb = bf16(x)  (feeds the full-TDM GEMM).
// ---------------------------------------------------------------------------
__launch_bounds__(256)
__global__ void lora_pre_kernel(const float* __restrict__ x,
                                const float* __restrict__ loraA,
                                float* __restrict__ xa,
                                float* __restrict__ rs,
                                __bf16* __restrict__ xb) {
  const int t  = threadIdx.x;
  const int m0 = blockIdx.x * 16;
  const int mr = t >> 4;
  const int r  = t & 15;

  __shared__ __align__(16) float xs[16][260];

  float acc  = 0.0f;
  float rsum = 0.0f;

  for (int kc = 0; kc < kK; kc += 256) {
    {
      const int srow = t >> 4;
      const int scol = (t & 15) * 16;
      const float4* p = (const float4*)(x + (size_t)(m0 + srow) * kK + kc + scol);
      float4 v0 = p[0], v1 = p[1], v2 = p[2], v3 = p[3];
      *(float4*)&xs[srow][scol + 0]  = v0;
      *(float4*)&xs[srow][scol + 4]  = v1;
      *(float4*)&xs[srow][scol + 8]  = v2;
      *(float4*)&xs[srow][scol + 12] = v3;
      if (xb) {   // emit bf16 copy of x for the TDM GEMM A-operand
        float af[16];
        *(float4*)&af[0]  = v0;
        *(float4*)&af[4]  = v1;
        *(float4*)&af[8]  = v2;
        *(float4*)&af[12] = v3;
        v8bf h0, h1;
#pragma unroll
        for (int j = 0; j < 8; ++j) {
          h0[j] = (__bf16)af[j];
          h1[j] = (__bf16)af[8 + j];
        }
        __bf16* xo = xb + (size_t)(m0 + srow) * kK + kc + scol;
        *(v8bf*)(xo)     = h0;
        *(v8bf*)(xo + 8) = h1;
      }
    }
    __syncthreads();
#pragma unroll 8
    for (int k = 0; k < 256; ++k) {
      const float xv = xs[mr][k];
      acc  += xv * loraA[(size_t)(kc + k) * kR + r];
      rsum += (float)(__bf16)xv;
    }
    __syncthreads();
  }
  xa[(size_t)(m0 + mr) * kR + r] = acc;
  if (r == 0) rs[m0 + mr] = rsum;
}

// ---------------------------------------------------------------------------
// TDM helper: one tensor_load_to_lds for a 128x32 bf16 tile out of a
// row-major [rows][4096] bf16 tensor, with 16B LDS padding per 64B row
// (reproduces the [128][40]-half padded LDS layout in hardware).
// ---------------------------------------------------------------------------
#if ATHENA_HAVE_TDM
__device__ __forceinline__ void tdm_load_tile(uint64_t gaddr, uint32_t ldsaddr) {
  const u32x4 g0 = {
      1u,                                     // count=1, user D#, no gather
      ldsaddr,                                // LDS byte address of tile
      (uint32_t)gaddr,                        // global_addr[31:0]
      ((uint32_t)(gaddr >> 32) & 0x01FFFFFFu) // global_addr[56:32]
          | (2u << 30)                        // type = 2 ("image")
  };
  const i32x8 g1 = {
      (1 << 16) | (1 << 20) | (3 << 22) | (3 << 25), // data_size=2B, pad en,
                                                     // 16DW interval, 4DW pad
      (int)(4096u << 16),   // tensor_dim0[15:0] = 4096
      (int)(4096u << 16),   // tensor_dim1[15:0] = 4096
      (int)(32u << 16),     // tile_dim0 = 32
      128,                  // tile_dim1 = 128
      4096,                 // tensor_dim0_stride[31:0]
      0, 0
  };
  const i32x4 z4 = {0, 0, 0, 0};
#if __clang_major__ >= 23
  const i32x8 z8 = {0, 0, 0, 0, 0, 0, 0, 0};
  __builtin_amdgcn_tensor_load_to_lds(g0, g1, z4, z4, z8, 0);
#else
  __builtin_amdgcn_tensor_load_to_lds(g0, g1, z4, z4, 0);
#endif
}

// ---------------------------------------------------------------------------
// Kernel 2 (TDM): WMMA GEMM  acc = x_bf16 @ qb^T.
//   ATDM=true : both A and B tiles via Tensor Data Mover (pure ds+wmma loop)
//   ATDM=false: A via register path from fp32 x, B via TDM
// ---------------------------------------------------------------------------
template <bool ATDM>
__launch_bounds__(256)
__global__ void qgemm_lora_tdm_kernel(const float* __restrict__ x,
                                      const __bf16* __restrict__ xb,
                                      const __bf16* __restrict__ qb,
                                      const float* __restrict__ scale,
                                      const float* __restrict__ zp,
                                      const float* __restrict__ bias,
                                      const float* __restrict__ loraB,
                                      const float* __restrict__ xa,
                                      const float* __restrict__ rs,
                                      __hip_bfloat16* __restrict__ out) {
  constexpr int BM = 128, BN = 128, BK = 32;
  constexpr int LDH = 40;
  constexpr int KT  = kK / BK;

  __shared__ __align__(16) __bf16 As[2][BM][LDH];
  __shared__ __align__(16) __bf16 Bs[2][BN][LDH];
  __shared__ __align__(16) float  xa_s[BM][kR];
  __shared__ float lb_s[BN][kR + 1];
  __shared__ float rs_s[BM];
  __shared__ float sc_s[BN];
  __shared__ float zp_s[BN];
  __shared__ float bi_s[BN];

  const int t    = threadIdx.x;
  const int lane = t & 31;
  const int wid  = t >> 5;
  const int wm   = wid & 3;
  const int wn   = wid >> 2;
  const int g    = lane >> 4;
  const int l16  = lane & 15;

  const int m0 = blockIdx.y * BM;
  const int n0 = blockIdx.x * BN;

  // ---- stage per-block epilogue data ----
  {
    const int rrow = t >> 1;
    const int half = t & 1;
    const float4* xap = (const float4*)(xa + (size_t)(m0 + rrow) * kR + half * 8);
    float4 q0 = xap[0], q1 = xap[1];
    *(float4*)&xa_s[rrow][half * 8 + 0] = q0;
    *(float4*)&xa_s[rrow][half * 8 + 4] = q1;
#pragma unroll
    for (int rr = 0; rr < 8; ++rr)
      lb_s[rrow][half * 8 + rr] = loraB[(size_t)(half * 8 + rr) * kN + n0 + rrow];
    if (t < BM) rs_s[t] = rs[m0 + t];
    if (t < BN) {
      sc_s[t] = scale[n0 + t];
      zp_s[t] = zp[n0 + t];
      bi_s[t] = bias[n0 + t];
    }
  }

  v8f acc[2][4];
  {
    const v8f zacc = {0.f, 0.f, 0.f, 0.f, 0.f, 0.f, 0.f, 0.f};
#pragma unroll
    for (int i = 0; i < 2; ++i)
#pragma unroll
      for (int j = 0; j < 4; ++j) acc[i][j] = zacc;
  }

  // ---- A register path state (only used when !ATDM) ----
  const int trow = t >> 1;
  const int tko  = (t & 1) * 16;
  const float* xg = x + (size_t)(m0 + trow) * kK + tko;
  float4 ra[4];
  if (!ATDM) {
#pragma unroll
    for (int i = 0; i < 4; ++i) ra[i] = ((const float4*)xg)[i];
  }

  // ---- TDM descriptors (uniform); wave0 drives A, wave1 drives B ----
  const uint64_t arow_bytes = (uint64_t)(uintptr_t)(xb + (size_t)m0 * kK);
  const uint64_t qrow_bytes = (uint64_t)(uintptr_t)(qb + (size_t)n0 * kK);
  uint32_t as_lds[2], bs_lds[2];
  as_lds[0] = (uint32_t)(uintptr_t)&As[0][0][0];  // low 32 bits = LDS offset
  as_lds[1] = (uint32_t)(uintptr_t)&As[1][0][0];
  bs_lds[0] = (uint32_t)(uintptr_t)&Bs[0][0][0];
  bs_lds[1] = (uint32_t)(uintptr_t)&Bs[1][0][0];

  if (ATDM && wid == 0) tdm_load_tile(arow_bytes, as_lds[0]);   // A tile 0
  if (wid == 1)         tdm_load_tile(qrow_bytes, bs_lds[0]);   // B tile 0

  for (int kt = 0; kt < KT; ++kt) {
    const int buf = kt & 1;

    if (!ATDM) {
      // store A tile (convert fp32 -> bf16) into LDS
      float af[16];
      *(float4*)&af[0]  = ra[0];
      *(float4*)&af[4]  = ra[1];
      *(float4*)&af[8]  = ra[2];
      *(float4*)&af[12] = ra[3];
      v8bf h0, h1;
#pragma unroll
      for (int i = 0; i < 8; ++i) {
        h0[i] = (__bf16)af[i];
        h1[i] = (__bf16)af[8 + i];
      }
      *(v8bf*)&As[buf][trow][tko + 0] = h0;
      *(v8bf*)&As[buf][trow][tko + 8] = h1;
      if (kt + 1 < KT) {
        const float* xp = xg + (size_t)(kt + 1) * BK;
#pragma unroll
        for (int i = 0; i < 4; ++i) ra[i] = ((const float4*)xp)[i];
      }
    }

    // current tiles' DMAs must be complete before anyone reads them.
    // Unguarded: trivially satisfied on waves that issued nothing.
    __builtin_amdgcn_s_wait_tensorcnt(0);
    __syncthreads();

    // issue next tiles AFTER the barrier (no wave can still be reading the
    // destination buffer); the DMAs overlap this iteration's WMMAs.
    if (kt + 1 < KT) {
      if (ATDM && wid == 0)
        tdm_load_tile(arow_bytes + (size_t)(kt + 1) * BK * 2,
                      as_lds[(kt + 1) & 1]);
      if (wid == 1)
        tdm_load_tile(qrow_bytes + (size_t)(kt + 1) * BK * 2,
                      bs_lds[(kt + 1) & 1]);
    }

    // fragments + 8 WMMAs
    v16bf a[2], b[4];
#pragma unroll
    for (int mf = 0; mf < 2; ++mf) {
      const __bf16* p = &As[buf][32 * wm + 16 * mf + l16][0];
      v8bf c0 = *(const v8bf*)(p + g * 8);
      v8bf c1 = *(const v8bf*)(p + 16 + g * 8);
      a[mf] = __builtin_shufflevector(c0, c1,
                                      0, 1, 2, 3, 4, 5, 6, 7,
                                      8, 9, 10, 11, 12, 13, 14, 15);
    }
#pragma unroll
    for (int nf = 0; nf < 4; ++nf) {
      const __bf16* p = &Bs[buf][64 * wn + 16 * nf + l16][0];
      v8bf c0 = *(const v8bf*)(p + g * 8);
      v8bf c1 = *(const v8bf*)(p + 16 + g * 8);
      b[nf] = __builtin_shufflevector(c0, c1,
                                      0, 1, 2, 3, 4, 5, 6, 7,
                                      8, 9, 10, 11, 12, 13, 14, 15);
    }
#pragma unroll
    for (int mf = 0; mf < 2; ++mf)
#pragma unroll
      for (int nf = 0; nf < 4; ++nf)
        acc[mf][nf] = __builtin_amdgcn_wmma_f32_16x16x32_bf16(
            false, a[mf], false, b[nf], (short)0, acc[mf][nf], false, false);
  }

  // ---- epilogue: dequant correction + bias + LoRA, bf16 store ----
#pragma unroll
  for (int nf = 0; nf < 4; ++nf) {
    const int nl  = 64 * wn + 16 * nf + l16;
    const float sc = sc_s[nl];
    const float z  = zp_s[nl];
    const float bi = bi_s[nl];
    float lb[kR];
#pragma unroll
    for (int r = 0; r < kR; ++r) lb[r] = lb_s[nl][r];
#pragma unroll
    for (int mf = 0; mf < 2; ++mf) {
      const v8f a8 = acc[mf][nf];
#pragma unroll
      for (int v = 0; v < 8; ++v) {
        const int ml = 32 * wm + 16 * mf + 8 * g + v;
        float lora = 0.0f;
#pragma unroll
        for (int r = 0; r < kR; ++r) lora += xa_s[ml][r] * lb[r];
        const float val = sc * (a8[v] - z * rs_s[ml]) + bi + kAlpha * lora;
        out[(size_t)(m0 + ml) * kN + (n0 + nl)] = __float2bfloat16(val);
      }
    }
  }
}
#endif  // ATHENA_HAVE_TDM

// ---------------------------------------------------------------------------
// Kernel 2 (fallback): in-kernel int8->bf16 conversion,
// register-path double-buffered pipeline. Used if the TDM builtin is missing
// or the workspace is too small.
// ---------------------------------------------------------------------------
__launch_bounds__(256)
__global__ void qgemm_lora_kernel(const float* __restrict__ x,
                                  const int8_t* __restrict__ qw,
                                  const float* __restrict__ scale,
                                  const float* __restrict__ zp,
                                  const float* __restrict__ bias,
                                  const float* __restrict__ loraB,
                                  const float* __restrict__ xa,
                                  const float* __restrict__ rs,
                                  __hip_bfloat16* __restrict__ out) {
  constexpr int BM = 128, BN = 128, BK = 32;
  constexpr int LDH = 40;
  constexpr int KT  = kK / BK;

  __shared__ __align__(16) __bf16 As[2][BM][LDH];
  __shared__ __align__(16) __bf16 Bs[2][BN][LDH];
  __shared__ __align__(16) float  xa_s[BM][kR];
  __shared__ float lb_s[BN][kR + 1];
  __shared__ float rs_s[BM];
  __shared__ float sc_s[BN];
  __shared__ float zp_s[BN];
  __shared__ float bi_s[BN];

  const int t    = threadIdx.x;
  const int lane = t & 31;
  const int wid  = t >> 5;
  const int wm   = wid & 3;
  const int wn   = wid >> 2;
  const int g    = lane >> 4;
  const int l16  = lane & 15;

  const int m0 = blockIdx.y * BM;
  const int n0 = blockIdx.x * BN;

  {
    const int rrow = t >> 1;
    const int half = t & 1;
    const float4* xap = (const float4*)(xa + (size_t)(m0 + rrow) * kR + half * 8);
    float4 q0 = xap[0], q1 = xap[1];
    *(float4*)&xa_s[rrow][half * 8 + 0] = q0;
    *(float4*)&xa_s[rrow][half * 8 + 4] = q1;
#pragma unroll
    for (int rr = 0; rr < 8; ++rr)
      lb_s[rrow][half * 8 + rr] = loraB[(size_t)(half * 8 + rr) * kN + n0 + rrow];
    if (t < BM) rs_s[t] = rs[m0 + t];
    if (t < BN) {
      sc_s[t] = scale[n0 + t];
      zp_s[t] = zp[n0 + t];
      bi_s[t] = bias[n0 + t];
    }
  }

  v8f acc[2][4];
  {
    const v8f zacc = {0.f, 0.f, 0.f, 0.f, 0.f, 0.f, 0.f, 0.f};
#pragma unroll
    for (int i = 0; i < 2; ++i)
#pragma unroll
      for (int j = 0; j < 4; ++j) acc[i][j] = zacc;
  }

  const int trow = t >> 1;
  const int tko  = (t & 1) * 16;
  const float*  xg = x  + (size_t)(m0 + trow) * kK + tko;
  const int8_t* qg = qw + (size_t)(n0 + trow) * kK + tko;

  float4 ra[4];
  int4   rb;
#pragma unroll
  for (int i = 0; i < 4; ++i) ra[i] = ((const float4*)xg)[i];
  rb = *(const int4*)qg;

  for (int kt = 0; kt < KT; ++kt) {
    const int buf = kt & 1;
    {
      float af[16];
      *(float4*)&af[0]  = ra[0];
      *(float4*)&af[4]  = ra[1];
      *(float4*)&af[8]  = ra[2];
      *(float4*)&af[12] = ra[3];
      v8bf h0, h1;
#pragma unroll
      for (int i = 0; i < 8; ++i) {
        h0[i] = (__bf16)af[i];
        h1[i] = (__bf16)af[8 + i];
      }
      *(v8bf*)&As[buf][trow][tko + 0] = h0;
      *(v8bf*)&As[buf][trow][tko + 8] = h1;

      union { int4 v; int8_t b[16]; } u;
      u.v = rb;
      v8bf q0, q1;
#pragma unroll
      for (int i = 0; i < 8; ++i) {
        q0[i] = (__bf16)(float)u.b[i];
        q1[i] = (__bf16)(float)u.b[8 + i];
      }
      *(v8bf*)&Bs[buf][trow][tko + 0] = q0;
      *(v8bf*)&Bs[buf][trow][tko + 8] = q1;
    }
    if (kt + 1 < KT) {
      const float*  xp = xg + (size_t)(kt + 1) * BK;
      const int8_t* qp = qg + (size_t)(kt + 1) * BK;
#pragma unroll
      for (int i = 0; i < 4; ++i) ra[i] = ((const float4*)xp)[i];
      rb = *(const int4*)qp;
    }
    __syncthreads();

    v16bf a[2], b[4];
#pragma unroll
    for (int mf = 0; mf < 2; ++mf) {
      const __bf16* p = &As[buf][32 * wm + 16 * mf + l16][0];
      v8bf c0 = *(const v8bf*)(p + g * 8);
      v8bf c1 = *(const v8bf*)(p + 16 + g * 8);
      a[mf] = __builtin_shufflevector(c0, c1,
                                      0, 1, 2, 3, 4, 5, 6, 7,
                                      8, 9, 10, 11, 12, 13, 14, 15);
    }
#pragma unroll
    for (int nf = 0; nf < 4; ++nf) {
      const __bf16* p = &Bs[buf][64 * wn + 16 * nf + l16][0];
      v8bf c0 = *(const v8bf*)(p + g * 8);
      v8bf c1 = *(const v8bf*)(p + 16 + g * 8);
      b[nf] = __builtin_shufflevector(c0, c1,
                                      0, 1, 2, 3, 4, 5, 6, 7,
                                      8, 9, 10, 11, 12, 13, 14, 15);
    }
#pragma unroll
    for (int mf = 0; mf < 2; ++mf)
#pragma unroll
      for (int nf = 0; nf < 4; ++nf)
        acc[mf][nf] = __builtin_amdgcn_wmma_f32_16x16x32_bf16(
            false, a[mf], false, b[nf], (short)0, acc[mf][nf], false, false);
  }

#pragma unroll
  for (int nf = 0; nf < 4; ++nf) {
    const int nl  = 64 * wn + 16 * nf + l16;
    const float sc = sc_s[nl];
    const float z  = zp_s[nl];
    const float bi = bi_s[nl];
    float lb[kR];
#pragma unroll
    for (int r = 0; r < kR; ++r) lb[r] = lb_s[nl][r];
#pragma unroll
    for (int mf = 0; mf < 2; ++mf) {
      const v8f a8 = acc[mf][nf];
#pragma unroll
      for (int v = 0; v < 8; ++v) {
        const int ml = 32 * wm + 16 * mf + 8 * g + v;
        float lora = 0.0f;
#pragma unroll
        for (int r = 0; r < kR; ++r) lora += xa_s[ml][r] * lb[r];
        const float val = sc * (a8[v] - z * rs_s[ml]) + bi + kAlpha * lora;
        out[(size_t)(m0 + ml) * kN + (n0 + nl)] = __float2bfloat16(val);
      }
    }
  }
}

// ---------------------------------------------------------------------------
extern "C" void kernel_launch(void* const* d_in, const int* in_sizes, int n_in,
                              void* d_out, int out_size, void* d_ws, size_t ws_size,
                              hipStream_t stream) {
  (void)in_sizes; (void)n_in; (void)out_size;

  const float*   x     = (const float*)d_in[0];
  const int8_t*  qw    = (const int8_t*)d_in[1];
  const float*   scale = (const float*)d_in[2];
  const float*   zp    = (const float*)d_in[3];
  const float*   bias  = (const float*)d_in[4];
  const float*   loraA = (const float*)d_in[5];
  const float*   loraB = (const float*)d_in[6];
  __hip_bfloat16* out  = (__hip_bfloat16*)d_out;

#if ATHENA_HAVE_TDM
  const size_t xb_bytes = (size_t)kM * kK * 2;                       // 128 MB
  const size_t qb_bytes = (size_t)kN * kK * 2;                       //  32 MB
  const size_t aux_bytes = ((size_t)kM * kR + kM) * 4;               // 1.06 MB
  const dim3 grid2(kN / 128, kM / 128);

  if (ws_size >= xb_bytes + qb_bytes + aux_bytes) {
    // Full TDM: both operands DMA'd; GEMM loop is pure ds_load + wmma.
    __bf16* xb = (__bf16*)d_ws;
    __bf16* qb = (__bf16*)((char*)d_ws + xb_bytes);
    float*  xa = (float*)((char*)d_ws + xb_bytes + qb_bytes);
    float*  rs = xa + (size_t)kM * kR;
    convert_q_kernel<<<(kN * kK) / (16 * 256), 256, 0, stream>>>(qw, qb);
    lora_pre_kernel<<<kM / 16, 256, 0, stream>>>(x, loraA, xa, rs, xb);
    qgemm_lora_tdm_kernel<true><<<grid2, 256, 0, stream>>>(
        x, xb, qb, scale, zp, bias, loraB, xa, rs, out);
    return;
  }
  if (ws_size >= qb_bytes + aux_bytes) {
    // B via TDM, A via register path.
    __bf16* qb = (__bf16*)d_ws;
    float*  xa = (float*)((char*)d_ws + qb_bytes);
    float*  rs = xa + (size_t)kM * kR;
    convert_q_kernel<<<(kN * kK) / (16 * 256), 256, 0, stream>>>(qw, qb);
    lora_pre_kernel<<<kM / 16, 256, 0, stream>>>(x, loraA, xa, rs, nullptr);
    qgemm_lora_tdm_kernel<false><<<grid2, 256, 0, stream>>>(
        x, nullptr, qb, scale, zp, bias, loraB, xa, rs, out);
    return;
  }
#endif
  float* xa = (float*)d_ws;
  float* rs = xa + (size_t)kM * kR;
  lora_pre_kernel<<<kM / 16, 256, 0, stream>>>(x, loraA, xa, rs, nullptr);
  qgemm_lora_kernel<<<dim3(kN / 128, kM / 128), 256, 0, stream>>>(
      x, qw, scale, zp, bias, loraB, xa, rs, out);
}
